// TopKAuxSAE_39187281609290
// MI455X (gfx1250) — matrix operand order repaired and checked by reading, not compiled, not run on hardware
//
#include <hip/hip_runtime.h>

#define BATCH_N 16384
#define DMODEL 1024
#define NFEAT 8192
#define TOPK 32

typedef __attribute__((ext_vector_type(16))) __bf16 v16bf;
typedef __attribute__((ext_vector_type(8)))  float  v8f;
typedef __attribute__((ext_vector_type(4)))  float  v4f;
typedef __attribute__((ext_vector_type(4)))  unsigned int v4u;
typedef __attribute__((ext_vector_type(2)))  unsigned int v2u;

struct Frag32B { v4u lo, hi; };

// Pack two f32 -> two bf16 (low/high) cheaply:
// one v_add_u32 per element (round-half-up) + one v_perm_b32 per pair.
__device__ __forceinline__ unsigned int pack2bf(float a, float b) {
  const unsigned int ua = __float_as_uint(a) + 0x8000u;
  const unsigned int ub = __float_as_uint(b) + 0x8000u;
  // result bytes [3:2] = ub[3:2], [1:0] = ua[3:2]  -> {bf16(b), bf16(a)}
  return __builtin_amdgcn_perm(ub, ua, 0x07060302u);
}

// ---------------------------------------------------------------------------
// Kernel 1: pre = (x - b_dec) @ W_enc + b_enc, written into the f region.
// bf16 WMMA 16x16x32, 128x128 block tile, 8 waves (2x4), 64x32 per wave.
// ---------------------------------------------------------------------------
__global__ __launch_bounds__(256) void sae_enc_kernel(
    const float* __restrict__ x, const float* __restrict__ Wenc,
    const float* __restrict__ benc, const float* __restrict__ bdec,
    float* __restrict__ pre)
{
  __shared__ alignas(16) unsigned short ldsX[128 * 32]; // [M][K] bf16, K-contig
  __shared__ alignas(16) unsigned short ldsW[128 * 32]; // [N][K] bf16 (transposed)

  const int tid  = threadIdx.x;
  const int lane = tid & 31;
  const int wid  = tid >> 5;
  const int wm = (wid & 1) * 64;   // wave M offset in block tile
  const int wn = (wid >> 1) * 32;  // wave N offset in block tile
  const int mbase = blockIdx.y * 128;
  const int nbase = blockIdx.x * 128;
  const int lr = lane & 15;        // row-in-16
  const int lh = lane >> 4;        // half select

  // global-load assignments
  const int xr = tid >> 3;         // 0..31  (X row group)
  const int xc = (tid & 7) * 4;    // 0..28  (X k offset, float4)
  const int wr = tid >> 5;         // 0..7   (W k-row group)
  const int wc = (tid & 31) * 4;   // 0..124 (W n offset, float4)

  v8f acc[4][2];
#pragma unroll
  for (int m = 0; m < 4; ++m)
#pragma unroll
    for (int n = 0; n < 2; ++n)
#pragma unroll
      for (int v = 0; v < 8; ++v) acc[m][n][v] = 0.0f;

  v4f xs[4], ws[4], bd;

  auto loadg = [&](int kk) {
    const int k0 = kk * 32;
    bd = *(const v4f*)(bdec + k0 + xc);
#pragma unroll
    for (int i = 0; i < 4; ++i)
      xs[i] = *(const v4f*)(x + (size_t)(mbase + xr + i * 32) * DMODEL + k0 + xc);
#pragma unroll
    for (int i = 0; i < 4; ++i)
      ws[i] = *(const v4f*)(Wenc + (size_t)(k0 + wr + i * 8) * NFEAT + nbase + wc);
  };

  auto stlds = [&]() {
#pragma unroll
    for (int i = 0; i < 4; ++i) {
      const int row = xr + i * 32;
      v2u pk;
      pk.x = pack2bf(xs[i].x - bd.x, xs[i].y - bd.y);
      pk.y = pack2bf(xs[i].z - bd.z, xs[i].w - bd.w);
      *reinterpret_cast<v2u*>(&ldsX[row * 32 + xc]) = pk;
    }
#pragma unroll
    for (int i = 0; i < 4; ++i) {       // transpose W into [N][K]
      const int kr = wr + i * 8;
      const unsigned int p01 = pack2bf(ws[i].x, ws[i].y);
      const unsigned int p23 = pack2bf(ws[i].z, ws[i].w);
      ldsW[(wc + 0) * 32 + kr] = (unsigned short)(p01 & 0xffffu);
      ldsW[(wc + 1) * 32 + kr] = (unsigned short)(p01 >> 16);
      ldsW[(wc + 2) * 32 + kr] = (unsigned short)(p23 & 0xffffu);
      ldsW[(wc + 3) * 32 + kr] = (unsigned short)(p23 >> 16);
    }
  };

  loadg(0);
  for (int kk = 0; kk < DMODEL / 32; ++kk) {
    stlds();
    __syncthreads();
    if (kk + 1 < DMODEL / 32) loadg(kk + 1);   // overlap with WMMA below

    v16bf afr[4], bfr[2];
    // A frag (16-bit 16x32): lane<16 holds K0..7 / K16..23; lane>=16 K8..15 / K24..31
#pragma unroll
    for (int m = 0; m < 4; ++m) {
      const unsigned short* p = &ldsX[(wm + m * 16 + lr) * 32 + lh * 8];
      Frag32B fc;
      fc.lo = *(const v4u*)p;
      fc.hi = *(const v4u*)(p + 16);           // +32 bytes
      afr[m] = __builtin_bit_cast(v16bf, fc);
    }
    // B frag (16-bit 32x16, lane=N): lane<16 holds K0..7 / K16..23; lane>=16 K8..15 / K24..31
#pragma unroll
    for (int n = 0; n < 2; ++n) {
      const unsigned short* p = &ldsW[(wn + n * 16 + lr) * 32 + lh * 8];
      Frag32B fc;
      fc.lo = *(const v4u*)p;
      fc.hi = *(const v4u*)(p + 16);           // +32 bytes
      bfr[n] = __builtin_bit_cast(v16bf, fc);
    }
#pragma unroll
    for (int m = 0; m < 4; ++m)
#pragma unroll
      for (int n = 0; n < 2; ++n)
        acc[m][n] = __builtin_amdgcn_wmma_f32_16x16x32_bf16(
            false, afr[m], false, bfr[n], (short)0, acc[m][n], false, false);
    __syncthreads();
  }

  // Epilogue: C layout — VGPR v, lanes 0-15: M=v,N=lane; lanes 16-31: M=v+8,N=lane-16
#pragma unroll
  for (int n = 0; n < 2; ++n) {
    const int gcol = nbase + wn + n * 16 + lr;
    const float bc = benc[gcol];
#pragma unroll
    for (int m = 0; m < 4; ++m) {
#pragma unroll
      for (int v = 0; v < 8; ++v) {
        const int grow = mbase + wm + m * 16 + lh * 8 + v;
        pre[(size_t)grow * NFEAT + gcol] = acc[m][n][v] + bc;
      }
    }
  }
}

// ---------------------------------------------------------------------------
// Kernel 2: per-row exact top-K (radix select over monotone keys), ReLU-masked
// in-place writeback of f, then fused sparse decode recon = f @ W_dec + b_dec.
// One block (256 threads) per row; keys in registers; all vmem is b128.
// Thread t owns elements { j*1024 + t*4 + e : j in 0..7, e in 0..3 }.
// ---------------------------------------------------------------------------
__global__ __launch_bounds__(256) void sae_topk_dec_kernel(
    float* __restrict__ f, const float* __restrict__ Wdec,
    const float* __restrict__ bdec, float* __restrict__ recon)
{
  const int row = blockIdx.x;
  const int tid = threadIdx.x;
  float* frow = f + (size_t)row * NFEAT;

  unsigned int keys[32];
  float vals[32];
#pragma unroll
  for (int j = 0; j < 8; ++j) {
    const v4f v = *(const v4f*)(frow + j * 1024 + tid * 4);
#pragma unroll
    for (int e = 0; e < 4; ++e) {
      vals[j * 4 + e] = v[e];
      const unsigned int u = __float_as_uint(v[e]);
      keys[j * 4 + e] = (u & 0x80000000u) ? ~u : (u | 0x80000000u); // monotone
    }
  }

  __shared__ int red[8];
  __shared__ int s_total;
  unsigned int prefix = 0, pmask = 0;
  int remaining = TOPK;

  for (int bit = 31; bit >= 0; --bit) {
    const unsigned int b = 1u << bit;
    const unsigned int tmask = pmask | b;
    const unsigned int tpref = prefix | b;
    int c = 0;
#pragma unroll
    for (int j = 0; j < 32; ++j) c += ((keys[j] & tmask) == tpref) ? 1 : 0;
#pragma unroll
    for (int off = 16; off > 0; off >>= 1) c += __shfl_down(c, off, 32);
    if ((tid & 31) == 0) red[tid >> 5] = c;
    __syncthreads();
    if (tid == 0) {
      int t = 0;
#pragma unroll
      for (int i = 0; i < 8; ++i) t += red[i];
      s_total = t;
    }
    __syncthreads();
    const int total = s_total;
    if (total >= remaining) prefix = tpref; else remaining -= total;
    pmask = tmask;
  }
  // prefix == key of the K-th largest; `remaining` = #ties admitted to top-K

  __shared__ int s_tie, s_cnt;
  __shared__ int   s_idx[TOPK];
  __shared__ float s_val[TOPK];
  if (tid == 0) { s_tie = 0; s_cnt = 0; }
  __syncthreads();
#pragma unroll
  for (int j = 0; j < 8; ++j) {
    v4f o;
#pragma unroll
    for (int e = 0; e < 4; ++e) {
      const int l = j * 4 + e;
      const int idx = j * 1024 + tid * 4 + e;
      bool keep = keys[l] > prefix;
      if (!keep && keys[l] == prefix) {
        const int oo = atomicAdd(&s_tie, 1);
        keep = (oo < remaining);
      }
      float out = 0.0f;
      if (keep) {
        out = vals[l] > 0.0f ? vals[l] : 0.0f;   // relu(top-k value)
        const int p = atomicAdd(&s_cnt, 1);
        s_idx[p] = idx;
        s_val[p] = out;
      }
      o[e] = out;
    }
    *(v4f*)(frow + j * 1024 + tid * 4) = o;
  }
  __syncthreads();

  // Fused sparse decode: W_dec is L2-resident (32 MB << 192 MB L2).
  // Thread t owns output cols [t*4, t*4+4) -> one b128 load per active feature.
  const int cnt = s_cnt;
  v4f a = *(const v4f*)(bdec + tid * 4);
  for (int j = 0; j < cnt; ++j) {
    const v4f w = *(const v4f*)(Wdec + (size_t)s_idx[j] * DMODEL + tid * 4);
    a += s_val[j] * w;
  }
  *(v4f*)(recon + (size_t)row * DMODEL + tid * 4) = a;
}

// ---------------------------------------------------------------------------
extern "C" void kernel_launch(void* const* d_in, const int* in_sizes, int n_in,
                              void* d_out, int out_size, void* d_ws, size_t ws_size,
                              hipStream_t stream) {
  const float* x    = (const float*)d_in[0];
  const float* Wenc = (const float*)d_in[1];
  const float* benc = (const float*)d_in[2];
  const float* Wdec = (const float*)d_in[3];
  const float* bdec = (const float*)d_in[4];

  float* recon = (float*)d_out;                                  // [B, d_model]
  float* fbuf  = (float*)d_out + (size_t)BATCH_N * DMODEL;       // [B, n_features]

  dim3 g1(NFEAT / 128, BATCH_N / 128);   // (64, 128) tiles
  sae_enc_kernel<<<g1, 256, 0, stream>>>(x, Wenc, benc, bdec, fbuf);
  sae_topk_dec_kernel<<<BATCH_N, 256, 0, stream>>>(fbuf, Wdec, bdec, recon);
}
